// GraphiT_Layer_17549236372058
// MI455X (gfx1250) — matrix operand order repaired and checked by reading, not compiled, not run on hardware
//
#include <hip/hip_runtime.h>
#include <hip/hip_bf16.h>
#include <math.h>

// Problem constants (from reference)
#define BB 4
#define NN 256
#define IN_DIM 128
#define IN_DIM_E 64
#define HD 256          // H*D = 8*32
#define TJ 16           // j-chunk rows per iteration
#define NCHUNK (NN / TJ)

// Branchless masking constants: masked scores get MASK_NEG; running max starts
// at INIT_M > MASK_NEG so exp(MASK_NEG - m) underflows to exactly 0 and an
// all-masked column finishes with lsum == 0 (-> output 0). All finite, no NaNs.
#define MASK_NEG (-3.0e38f)
#define INIT_M   (-1.0e38f)

typedef __attribute__((ext_vector_type(16))) __bf16 v16bf;
typedef __attribute__((ext_vector_type(8)))  float  v8f;

union Frag {
    v16bf v;
    unsigned int d[8];
    unsigned short u[16];
};

// round-to-nearest-even float -> bf16 bits
static __device__ inline unsigned short f2bf(float f) {
    unsigned u = __float_as_uint(f);
    u += 0x7FFFu + ((u >> 16) & 1u);
    return (unsigned short)(u >> 16);
}

// ---------------------------------------------------------------------------
// Kernel 1: Q/K/V projections. One block per (b,i) row; thread c computes
// output channel c for Q, K (pre-scaled by D^-0.5) and V. 0.2 GFLOP total.
// ---------------------------------------------------------------------------
__global__ __launch_bounds__(256) void qkv_kernel(
    const float* __restrict__ h,
    const float* __restrict__ Wq, const float* __restrict__ Wk,
    const float* __restrict__ Wv,
    float* __restrict__ Qb, float* __restrict__ Kb, float* __restrict__ Vb)
{
    __shared__ float sH[IN_DIM];
    const int row = blockIdx.x;       // b*N + i
    const int c   = threadIdx.x;
    if (c < IN_DIM) sH[c] = h[(size_t)row * IN_DIM + c];
    __syncthreads();

    float q = 0.f, k = 0.f, v = 0.f;
    #pragma unroll 4
    for (int t = 0; t < IN_DIM; ++t) {
        const float hv = sH[t];
        q = fmaf(hv, Wq[(size_t)t * HD + c], q);
        k = fmaf(hv, Wk[(size_t)t * HD + c], k);
        v = fmaf(hv, Wv[(size_t)t * HD + c], v);
    }
    const size_t o = (size_t)row * HD + c;
    Qb[o] = q;
    Kb[o] = k * 0.17677669529663687f;   // D^-0.5, D=32
    Vb[o] = v;
}

// ---------------------------------------------------------------------------
// Kernel 2: fused E-GEMM (bf16 WMMA) + scores + flash softmax over j + V mix.
// One 256-thread block (8 waves) per (b,i).
// ---------------------------------------------------------------------------
__global__ __launch_bounds__(256) void attn_kernel(
    const float* __restrict__ edge,   // [B,N,N,64]
    const float* __restrict__ mask,   // [B,N]
    const float* __restrict__ We,     // [64,256]
    const float* __restrict__ Qb, const float* __restrict__ Kb,
    const float* __restrict__ Vb,     // each [B*N,256]
    float* __restrict__ out)          // [B*N,256]
{
    // Padded LDS (row strides chosen so dword fragment loads are bank-clean)
    __shared__ unsigned short sWe[IN_DIM_E][HD + 2];   // bf16 B operand, 33 KB
    __shared__ unsigned short sA[TJ][IN_DIM_E + 2];    // bf16 A tile,     2 KB
    __shared__ float          sE[TJ][HD + 2];          // E tile,         16.1 KB
    __shared__ float          sQ[HD];
    __shared__ float          sQK[TJ][8];              // Q.K + mask bias
    __shared__ float          sM[NN];

    const int tid  = threadIdx.x;
    const int lane = tid & 31;
    const int wave = tid >> 5;
    const int bi   = blockIdx.x;      // b*N + i
    const int b    = bi >> 8;
    const int i    = bi & (NN - 1);

    // Stage We as bf16 (once per block; L2-resident source)
    for (int idx = tid; idx < IN_DIM_E * HD; idx += 256)
        sWe[idx >> 8][idx & (HD - 1)] = f2bf(We[idx]);
    sQ[tid] = Qb[(size_t)bi * HD + tid];
    sM[tid] = mask[b * NN + tid];
    __syncthreads();

    const float mi = sM[i];
    const int   hi = (lane >= 16) ? 1 : 0;   // lane half (WMMA layouts)
    const int   m  = lane & 15;
    const int   n0a = wave * 32;             // this wave's 32 output columns

    const float* edgeBase = edge + (size_t)bi * NN * IN_DIM_E;
    const float* vBase    = Vb + ((size_t)b * NN) * HD;

    // Flash state: thread owns column c = tid = h*32+d
    float mcur = INIT_M, lsum = 0.f, acc = 0.f;
    const int hcol = tid >> 5;

    for (int chunk = 0; chunk < NCHUNK; ++chunk) {
        const int j0 = chunk * TJ;

        // ---- stage edge tile (fp32 -> bf16), compute qk[j,h] (+mask bias)
        for (int idx = tid; idx < TJ * IN_DIM_E; idx += 256) {
            const int jl = idx >> 6, k = idx & 63;
            sA[jl][k] = f2bf(edgeBase[(size_t)(j0 + jl) * IN_DIM_E + k]);
        }
        if (tid < TJ * 8) {                       // waves 0..3: Q.K dots
            const int jl = tid >> 3, hh = tid & 7;
            const float* krow = Kb + ((size_t)(b * NN + j0 + jl)) * HD + hh * 32;
            const float* qrow = sQ + hh * 32;
            float s = 0.f;
            #pragma unroll
            for (int d = 0; d < 32; ++d) s = fmaf(qrow[d], krow[d], s);
            // fold pairwise mask into the additive bias (branchless hot loop)
            sQK[jl][hh] = (mi * sM[j0 + jl] != 0.f) ? s : MASK_NEG;
        }
        // prefetch next chunk's edge rows (global_prefetch_b8)
        if (wave == 7 && lane < TJ && chunk + 1 < NCHUNK)
            __builtin_prefetch(edgeBase + (size_t)(j0 + TJ + lane) * IN_DIM_E, 0, 0);
        __syncthreads();

        // ---- WMMA: E[16 x 32cols] per wave, K=64 as 2x k32 bf16 WMMAs
        // A frag (ISA 16-bit A 16x32): lane half selects K+8; vgpr r pairs K
        Frag a0, a1;
        #pragma unroll
        for (int r = 0; r < 8; ++r) {
            const int k0 = ((r < 4) ? 0 : 16) + hi * 8 + 2 * (r & 3);
            a0.d[r] = *(const unsigned int*)&sA[m][k0];
            a1.d[r] = *(const unsigned int*)&sA[m][32 + k0];
        }
        #pragma unroll
        for (int t = 0; t < 2; ++t) {
            const int n0 = n0a + t * 16;
            // B frag (32x16): col = lane&15, vgpr u packs K pair (same K order as A)
            Frag b0, b1;
            #pragma unroll
            for (int u = 0; u < 8; ++u) {
                const int kb = ((u < 4) ? 0 : 16) + hi * 8 + 2 * (u & 3);
                b0.u[2 * u]     = sWe[kb][n0 + m];
                b0.u[2 * u + 1] = sWe[kb + 1][n0 + m];
                b1.u[2 * u]     = sWe[32 + kb][n0 + m];
                b1.u[2 * u + 1] = sWe[32 + kb + 1][n0 + m];
            }
            v8f c = {};
            c = __builtin_amdgcn_wmma_f32_16x16x32_bf16(false, a0.v, false, b0.v,
                                                        (short)0, c, false, false);
            c = __builtin_amdgcn_wmma_f32_16x16x32_bf16(false, a1.v, false, b1.v,
                                                        (short)0, c, false, false);
            // D layout: vgpr r -> row r (+8 for upper lane half), col = lane&15
            #pragma unroll
            for (int r = 0; r < 8; ++r)
                sE[r + hi * 8][n0 + m] = c[r];
        }
        __syncthreads();

        // ---- branchless online softmax over this chunk's 16 keys + V mix.
        // Masked keys carry s ~= MASK_NEG < mcur, so p underflows to 0 and
        // contributes nothing; no divergence, V loads pipeline freely.
        #pragma unroll
        for (int jl = 0; jl < TJ; ++jl) {
            const float s    = sE[jl][tid] + sQK[jl][hcol];
            const float vv   = vBase[(size_t)(j0 + jl) * HD + tid];
            const float mnew = fmaxf(mcur, s);
            const float corr = __expf(mcur - mnew);
            const float p    = __expf(s - mnew);
            lsum = fmaf(lsum, corr, p);
            acc  = fmaf(acc,  corr, p * vv);
            mcur = mnew;
        }
        __syncthreads();   // protect sA/sQK/sE before next chunk rewrites
    }

    out[(size_t)bi * HD + tid] = (lsum > 0.f) ? (acc / lsum) : 0.f;
}

// ---------------------------------------------------------------------------
extern "C" void kernel_launch(void* const* d_in, const int* in_sizes, int n_in,
                              void* d_out, int out_size, void* d_ws, size_t ws_size,
                              hipStream_t stream) {
    (void)in_sizes; (void)n_in; (void)out_size; (void)ws_size;
    const float* h    = (const float*)d_in[0];
    const float* edge = (const float*)d_in[1];
    const float* mask = (const float*)d_in[2];
    const float* Wq   = (const float*)d_in[3];
    const float* Wk   = (const float*)d_in[4];
    const float* Wv   = (const float*)d_in[5];
    const float* We   = (const float*)d_in[6];
    float* out = (float*)d_out;

    // Workspace: Q, K(scaled), V  — 3 x 1 MB
    float* Qb = (float*)d_ws;
    float* Kb = Qb + (size_t)BB * NN * HD;
    float* Vb = Kb + (size_t)BB * NN * HD;

    qkv_kernel<<<BB * NN, 256, 0, stream>>>(h, Wq, Wk, Wv, Qb, Kb, Vb);
    attn_kernel<<<BB * NN, 256, 0, stream>>>(edge, mask, We, Qb, Kb, Vb, out);
}
